// pointnet2_cls_msg_29111288332586
// MI455X (gfx1250) — matrix operand order repaired and checked by reading, not compiled, not run on hardware
//
#include <hip/hip_runtime.h>
#include <hip/hip_bf16.h>
#include <cstdint>

// ---------------------------------------------------------------------------
// PointNet++ MSG classification forward for gfx1250 (MI455X).
// All MLP/FC matmuls run through v_wmma_f32_16x16x32_f16 (f16 in, f32 acc).
// Weights stream via global_load_b128 (explicit addrspace(1)), activations
// live in LDS and feed WMMA A-fragments via ds_load_b128.
// ---------------------------------------------------------------------------

typedef _Float16 h16;
typedef __attribute__((ext_vector_type(16))) _Float16 v16h;
typedef __attribute__((ext_vector_type(8)))  float    v8f;
typedef __attribute__((ext_vector_type(4)))  unsigned int u32x4;  // POD 16B

static constexpr int BB = 8;     // batch
static constexpr int NN = 2048;  // points

__device__ __forceinline__ v8f wmma_f16(v16h a, v16h b, v8f c) {
  // D = A(16x32) * B(32x16) + C, f32 accum
  return __builtin_amdgcn_wmma_f32_16x16x32_f16(false, a, false, b, (short)0, c,
                                                false, false);
}

union AF { v16h v; u32x4 q[2]; };

// 16-byte load through an explicit global (addrspace 1) pointer so the
// backend emits global_load_b128 instead of flat_load_b128.
typedef const u32x4 __attribute__((address_space(1)))* gb128_t;
__device__ __forceinline__ u32x4 ldg128(const h16* p) {
  return *(gb128_t)(const void*)p;
}

// A fragment (LDS/generic source): 16x32 f16, row-major, leading dim ld.
// lane<16 holds K {0..7, 16..23}; lane>=16 holds K {8..15, 24..31}.
__device__ __forceinline__ v16h frag_a_lds(const h16* base, int ld, int rowBase,
                                           int kb, int lane) {
  int row = rowBase + (lane & 15);
  int kh  = (lane >> 4) * 8;
  const h16* p = base + (size_t)row * ld + kb + kh;
  AF f;
  f.q[0] = *(const u32x4*)p;
  f.q[1] = *(const u32x4*)(p + 16);
  return f.v;
}

// A fragment from global memory.
__device__ __forceinline__ v16h frag_a_glb(const h16* base, int ld, int rowBase,
                                           int kb, int lane) {
  int row = rowBase + (lane & 15);
  int kh  = (lane >> 4) * 8;
  const h16* p = base + (size_t)row * ld + kb + kh;
  AF f;
  f.q[0] = ldg128(p);
  f.q[1] = ldg128(p + 16);
  return f.v;
}

// B fragment: 32x16 f16 from TRANSPOSED weights Wt[Cout][kd] (kd mult of 32),
// always resident in global memory.
// lane<16 -> K {0..7, 16..23} of column (lane&15); lane>=16 -> K {8..15, 24..31}.
__device__ __forceinline__ v16h frag_b_glb(const h16* wt, int kd, int colBase,
                                           int kb, int lane) {
  int col = colBase + (lane & 15);
  int kh  = (lane >> 4) * 8;
  const h16* p = wt + (size_t)col * kd + kb + kh;
  AF f;
  f.q[0] = ldg128(p);
  f.q[1] = ldg128(p + 16);
  return f.v;
}

// ------------------------- small utility kernels ---------------------------

__global__ void pn2_cvt_f16(const float* __restrict__ in, h16* __restrict__ out,
                            int n) {
  int t = blockIdx.x * blockDim.x + threadIdx.x;
  if (t < n) out[t] = (h16)in[t];
}

// W f32 (cin x cout, row-major) -> Wt f16 (coutPad x cinPad), zero padded.
__global__ void pn2_wconv(const float* __restrict__ W, h16* __restrict__ Wt,
                          int cin, int cout, int cinPad, int coutPad) {
  int t = blockIdx.x * blockDim.x + threadIdx.x;
  int n = coutPad * cinPad;
  if (t >= n) return;
  int o = t / cinPad, i = t - o * cinPad;
  float v = (o < cout && i < cin) ? W[(size_t)i * cout + o] : 0.f;
  Wt[t] = (h16)v;
}

// ------------------------------- FPS ---------------------------------------

__global__ void pn2_fps(const float* __restrict__ xyz, int Np, int M,
                        int* __restrict__ cents) {
  __shared__ float sxyz[NN * 3];
  __shared__ float sd[NN];
  __shared__ float rv[256];
  __shared__ int   ri[256];
  int b = blockIdx.x, tid = threadIdx.x;
  for (int e = tid; e < Np * 3; e += blockDim.x)
    sxyz[e] = xyz[(size_t)b * Np * 3 + e];
  for (int n = tid; n < Np; n += blockDim.x) sd[n] = 1e10f;
  __syncthreads();
  int cur = 0;
  for (int i = 0; i < M; ++i) {
    if (tid == 0) cents[b * M + i] = cur;
    float cx = sxyz[cur * 3 + 0], cy = sxyz[cur * 3 + 1], cz = sxyz[cur * 3 + 2];
    float bv = -1.f; int bi = 0x7fffffff;
    for (int n = tid; n < Np; n += blockDim.x) {
      float dx = sxyz[n * 3 + 0] - cx;
      float dy = sxyz[n * 3 + 1] - cy;
      float dz = sxyz[n * 3 + 2] - cz;
      float d = fminf(sd[n], dx * dx + dy * dy + dz * dz);
      sd[n] = d;
      if (d > bv) { bv = d; bi = n; }
    }
    rv[tid] = bv; ri[tid] = bi;
    __syncthreads();
    for (int st = 128; st > 0; st >>= 1) {
      if (tid < st) {
        if (rv[tid + st] > rv[tid] ||
            (rv[tid + st] == rv[tid] && ri[tid + st] < ri[tid])) {
          rv[tid] = rv[tid + st]; ri[tid] = ri[tid + st];
        }
      }
      __syncthreads();
    }
    cur = ri[0];
    __syncthreads();
  }
}

__global__ void pn2_gather_xyz(const float* __restrict__ xyz,
                               const int* __restrict__ idx, int Np, int M,
                               float* __restrict__ out) {
  int t = blockIdx.x * blockDim.x + threadIdx.x;
  if (t >= BB * M * 3) return;
  int i = t / 3, c = t - i * 3;
  int b = i / M;
  int gi = idx[i]; if (gi >= Np) gi = Np - 1; if (gi < 0) gi = 0;
  out[t] = xyz[((size_t)b * Np + gi) * 3 + c];
}

// --------------------------- Ball query (wave32) ---------------------------
// One wave per centroid; in-order ballot compaction reproduces the
// "K smallest indices within radius, padded with first" reference semantics.
__global__ void pn2_ball_query(const float* __restrict__ xyz,
                               const float* __restrict__ nxyz, int Np, int M,
                               int K, float r2, int* __restrict__ out) {
  int lane = threadIdx.x & 31;
  int g = (blockIdx.x * blockDim.x + threadIdx.x) >> 5;
  if (g >= BB * M) return;
  int b = g / M;
  float cx = nxyz[(size_t)g * 3 + 0];
  float cy = nxyz[(size_t)g * 3 + 1];
  float cz = nxyz[(size_t)g * 3 + 2];
  const float* base = xyz + (size_t)b * Np * 3;
  int cnt = 0, first = 0;
  for (int n0 = 0; n0 < Np; n0 += 32) {
    int n = n0 + lane;
    float dx = base[n * 3 + 0] - cx;
    float dy = base[n * 3 + 1] - cy;
    float dz = base[n * 3 + 2] - cz;
    float d2 = dx * dx + dy * dy + dz * dz;
    bool pred = d2 <= r2;
    unsigned mm = (unsigned)__ballot(pred);
    if (cnt == 0 && mm) first = n0 + (__ffs(mm) - 1);
    int pre = __popc(mm & ((1u << lane) - 1u));
    int pos = cnt + pre;
    if (pred && pos < K) out[(size_t)g * K + pos] = n;
    cnt += __popc(mm);
    if (cnt >= K) break;   // cnt is wave-uniform
  }
  for (int p = cnt + lane; p < K; p += 32) out[(size_t)g * K + p] = first;
}

// ---------------- Fused gather + 3-layer MLP + maxpool (per centroid) ------
// Block = one (b, m) group. Features & activations ping-pong entirely in LDS;
// each wave computes 16x16 WMMA tiles, weights stream from L2.
__global__ void pn2_group_mlp(
    const float* __restrict__ xyz, const h16* __restrict__ pfeat,
    const float* __restrict__ cxyz, const int* __restrict__ bq,
    int Np, int M, int K, int Cpts, int CinPad,
    const h16* wt0, const h16* wt1, const h16* wt2,
    const float* s0, const float* s1, const float* s2,
    const float* b0, const float* b1, const float* b2,
    int c0, int c1, int c2,
    h16* __restrict__ outp, int totC, int coff) {
  extern __shared__ float4 smemv[];
  char* smem = (char*)smemv;
  int* sidx = (int*)smem;
  size_t off = ((size_t)K * 4 + 15) & ~15ULL;
  float* scen = (float*)(smem + off); off += 16;
  h16* bufG = (h16*)(smem + off); off += ((size_t)K * CinPad * 2 + 15) & ~15ULL;
  int cA = c0 > c2 ? c0 : c2;
  h16* bufA = (h16*)(smem + off); off += ((size_t)K * cA * 2 + 15) & ~15ULL;
  h16* bufB = (h16*)(smem + off);

  int tid = threadIdx.x;
  int lane = tid & 31, wv = tid >> 5;
  int nwv = blockDim.x >> 5;
  int g = blockIdx.x, b = g / M;

  for (int k = tid; k < K; k += blockDim.x) sidx[k] = bq[(size_t)g * K + k];
  if (tid < 3) scen[tid] = cxyz[(size_t)g * 3 + tid];
  __syncthreads();

  // Gather: [K x CinPad] f16, channels = (xyz - center) ++ point feats, 0-pad.
  int tot = K * CinPad;
  for (int e = tid; e < tot; e += blockDim.x) {
    int k = e / CinPad, c = e - k * CinPad;
    int gi = sidx[k]; if (gi >= Np) gi = Np - 1; if (gi < 0) gi = 0;
    float v = 0.f;
    if (c < 3) v = xyz[((size_t)b * Np + gi) * 3 + c] - scen[c];
    else if (c < 3 + Cpts) v = (float)pfeat[((size_t)b * Np + gi) * Cpts + (c - 3)];
    bufG[e] = (h16)v;
  }
  __syncthreads();

  const h16* in = bufG;
  int ldin = CinPad;
  const h16*  wts[3] = {wt0, wt1, wt2};
  const float* ss[3] = {s0, s1, s2};
  const float* bs[3] = {b0, b1, b2};
  int cos[3] = {c0, c1, c2};

  for (int L = 0; L < 3; ++L) {
    int kd = ldin, co = cos[L];
    h16* ob = (L == 1) ? bufB : bufA;
    const h16* wt = wts[L];
    int ctiles = co >> 4;
    int tiles = (K >> 4) * ctiles;
    for (int t = wv; t < tiles; t += nwv) {
      int rt = t / ctiles, ct = t - rt * ctiles;
      v8f acc = {};
      for (int kb = 0; kb < kd; kb += 32)
        acc = wmma_f16(frag_a_lds(in, kd, rt << 4, kb, lane),
                       frag_b_glb(wt, kd, ct << 4, kb, lane), acc);
      int col = (ct << 4) + (lane & 15);
      float sc = ss[L][col], bi = bs[L][col];
      int rb = (rt << 4) + ((lane >> 4) << 3);
#pragma unroll
      for (int r = 0; r < 8; ++r) {
        float v = fmaxf(acc[r] * sc + bi, 0.f);
        ob[(size_t)(rb + r) * co + col] = (h16)v;
      }
    }
    __syncthreads();
    in = ob; ldin = co;
  }

  // max over the K group points -> concat output channel slot
  for (int c = tid; c < cos[2]; c += blockDim.x) {
    float mv = -3.4e38f;
    for (int k = 0; k < K; ++k) mv = fmaxf(mv, (float)in[(size_t)k * ldin + c]);
    outp[(size_t)g * totC + coff + c] = (h16)mv;
  }
}

// ------------------------- generic WMMA GEMM -------------------------------
// Y = act(X[rows x kd] * Wt^T * s + b); one 16x16 tile per wave.
__global__ void pn2_gemm16(const h16* __restrict__ X, const h16* __restrict__ Wt,
                           int kd, const float* __restrict__ s,
                           const float* __restrict__ bv, int relu,
                           h16* __restrict__ Yh, float* __restrict__ Yf, int ldy,
                           int rowTiles, int colTiles, int vRows, int vCols) {
  int lane = threadIdx.x & 31;
  int wv = (blockIdx.x * blockDim.x + threadIdx.x) >> 5;
  int tiles = rowTiles * colTiles;
  if (wv >= tiles) return;
  int rt = wv / colTiles, ct = wv - rt * colTiles;
  v8f acc = {};
  for (int kb = 0; kb < kd; kb += 32) {
    // hint next weight slab toward the caches while this one computes
    if (kb + 32 < kd)
      __builtin_prefetch(Wt + (size_t)(ct << 4) * kd + kb + 32, 0, 1);
    acc = wmma_f16(frag_a_glb(X, kd, rt << 4, kb, lane),
                   frag_b_glb(Wt, kd, ct << 4, kb, lane), acc);
  }
  int col = (ct << 4) + (lane & 15);
  float sc = (s && col < vCols) ? s[col] : 1.f;
  float bi = (bv && col < vCols) ? bv[col] : 0.f;
  int rb = (rt << 4) + ((lane >> 4) << 3);
#pragma unroll
  for (int r = 0; r < 8; ++r) {
    int row = rb + r;
    float v = acc[r] * sc + bi;
    if (relu) v = fmaxf(v, 0.f);
    if (row < vRows && col < vCols) {
      if (Yh) Yh[(size_t)row * ldy + col] = (h16)v;
      else    Yf[(size_t)row * ldy + col] = v;
    }
  }
}

// SA3 feature build: [B*128 x 672] = concat(new_xyz2, points2), zero padded.
__global__ void pn2_sa3_feat(const float* __restrict__ nxyz,
                             const h16* __restrict__ pts,
                             h16* __restrict__ out) {
  const int C = 672, R = BB * 128;
  int t = blockIdx.x * blockDim.x + threadIdx.x;
  if (t >= R * C) return;
  int row = t / C, c = t - row * C;
  float v = 0.f;
  if (c < 3) v = nxyz[row * 3 + c];
  else if (c < 643) v = (float)pts[(size_t)row * 640 + (c - 3)];
  out[t] = (h16)v;
}

// SA3 maxpool over 128 points -> FC input [16 x 1024] (rows 8..15 zero).
__global__ void pn2_sa3_pool(const h16* __restrict__ act, h16* __restrict__ fcin) {
  int t = blockIdx.x * blockDim.x + threadIdx.x;
  if (t >= 16 * 1024) return;
  int row = t >> 10, c = t & 1023;
  float v = 0.f;
  if (row < BB) {
    float mv = -3.4e38f;
    for (int k = 0; k < 128; ++k)
      mv = fmaxf(mv, (float)act[((size_t)(row * 128 + k) << 10) + c]);
    v = mv;
  }
  fcin[t] = (h16)v;
}

// ---------------------------------------------------------------------------

extern "C" void kernel_launch(void* const* d_in, const int* in_sizes, int n_in,
                              void* d_out, int out_size, void* d_ws,
                              size_t ws_size, hipStream_t stream) {
  (void)in_sizes; (void)n_in; (void)out_size; (void)ws_size;

  struct Lay { const float *W, *s, *b; int cin, cout, cinPad, coutPad; h16* wt; };
  auto pad16 = [](int x) { return (x + 15) & ~15; };
  auto pad32 = [](int x) { return (x + 31) & ~31; };

  int q = 0;
  const float* xyz = (const float*)d_in[q++];
  const float* pts = (const float*)d_in[q++];

  static const int sa1w[3][3] = {{32, 32, 64}, {64, 64, 128}, {64, 96, 128}};
  static const int sa2w[3][3] = {{64, 64, 128}, {128, 128, 256}, {128, 128, 256}};
  static const int sa3w[3] = {256, 512, 1024};

  Lay sa1[3][3], sa2[3][3], sa3l[3], fc[3];
  for (int sc = 0; sc < 3; ++sc) {
    int cin = 6;
    for (int l = 0; l < 3; ++l) {
      Lay& L = sa1[sc][l];
      L.W = (const float*)d_in[q++]; L.s = (const float*)d_in[q++];
      L.b = (const float*)d_in[q++];
      L.cin = cin; L.cout = sa1w[sc][l]; cin = L.cout;
    }
  }
  for (int sc = 0; sc < 3; ++sc) {
    int cin = 323;
    for (int l = 0; l < 3; ++l) {
      Lay& L = sa2[sc][l];
      L.W = (const float*)d_in[q++]; L.s = (const float*)d_in[q++];
      L.b = (const float*)d_in[q++];
      L.cin = cin; L.cout = sa2w[sc][l]; cin = L.cout;
    }
  }
  {
    int cin = 643;
    for (int l = 0; l < 3; ++l) {
      Lay& L = sa3l[l];
      L.W = (const float*)d_in[q++]; L.s = (const float*)d_in[q++];
      L.b = (const float*)d_in[q++];
      L.cin = cin; L.cout = sa3w[l]; cin = L.cout;
    }
  }
  fc[0].W = (const float*)d_in[q++]; fc[0].s = (const float*)d_in[q++];
  fc[0].b = (const float*)d_in[q++]; fc[0].cin = 1024; fc[0].cout = 512;
  fc[1].W = (const float*)d_in[q++]; fc[1].s = (const float*)d_in[q++];
  fc[1].b = (const float*)d_in[q++]; fc[1].cin = 512; fc[1].cout = 256;
  fc[2].W = (const float*)d_in[q++]; fc[2].s = nullptr;
  fc[2].b = (const float*)d_in[q++]; fc[2].cin = 256; fc[2].cout = 40;

  // workspace bump allocator (256B aligned)
  char* wsp = (char*)d_ws;
  auto alloc = [&](size_t bytes) -> void* {
    void* p = (void*)wsp;
    wsp += (bytes + 255) & ~(size_t)255;
    return p;
  };

  auto convW = [&](Lay& L) {
    L.cinPad = pad32(L.cin);
    L.coutPad = pad16(L.cout);
    L.wt = (h16*)alloc((size_t)L.cinPad * L.coutPad * sizeof(h16));
    int n = L.cinPad * L.coutPad;
    pn2_wconv<<<(n + 255) / 256, 256, 0, stream>>>(L.W, L.wt, L.cin, L.cout,
                                                   L.cinPad, L.coutPad);
  };
  for (int sc = 0; sc < 3; ++sc)
    for (int l = 0; l < 3; ++l) convW(sa1[sc][l]);
  for (int sc = 0; sc < 3; ++sc)
    for (int l = 0; l < 3; ++l) convW(sa2[sc][l]);
  for (int l = 0; l < 3; ++l) convW(sa3l[l]);
  for (int l = 0; l < 3; ++l) convW(fc[l]);

  (void)hipFuncSetAttribute((const void*)pn2_group_mlp,
                            hipFuncAttributeMaxDynamicSharedMemorySize,
                            200 * 1024);

  auto fusedShmem = [&](int K, int CinPad, int c0, int c1, int c2) -> size_t {
    size_t off = ((size_t)K * 4 + 15) & ~15ULL;
    off += 16;
    off += ((size_t)K * CinPad * 2 + 15) & ~15ULL;
    int cA = c0 > c2 ? c0 : c2;
    off += ((size_t)K * cA * 2 + 15) & ~15ULL;
    off += (size_t)K * c1 * 2;
    return off;
  };

  // points -> f16
  h16* pts_h = (h16*)alloc((size_t)BB * NN * 3 * sizeof(h16));
  pn2_cvt_f16<<<(BB * NN * 3 + 255) / 256, 256, 0, stream>>>(pts, pts_h,
                                                             BB * NN * 3);

  // ---------------- SA1: FPS 2048 -> 512, 3 scales ----------------
  int* c1i = (int*)alloc((size_t)BB * 512 * sizeof(int));
  pn2_fps<<<BB, 256, 0, stream>>>(xyz, NN, 512, c1i);
  float* nx1 = (float*)alloc((size_t)BB * 512 * 3 * sizeof(float));
  pn2_gather_xyz<<<(BB * 512 * 3 + 255) / 256, 256, 0, stream>>>(xyz, c1i, NN,
                                                                 512, nx1);
  h16* pts1 = (h16*)alloc((size_t)BB * 512 * 320 * sizeof(h16));
  {
    const float rad[3] = {0.1f, 0.2f, 0.4f};
    const int   kk[3]  = {16, 32, 128};
    int coff = 0;
    for (int sc = 0; sc < 3; ++sc) {
      int K = kk[sc];
      int* bq = (int*)alloc((size_t)BB * 512 * K * sizeof(int));
      pn2_ball_query<<<(BB * 512) / 8, 256, 0, stream>>>(
          xyz, nx1, NN, 512, K, rad[sc] * rad[sc], bq);
      Lay* L = sa1[sc];
      size_t sh = fusedShmem(K, 32, L[0].cout, L[1].cout, L[2].cout);
      pn2_group_mlp<<<BB * 512, 256, sh, stream>>>(
          xyz, pts_h, nx1, bq, NN, 512, K, 3, 32,
          L[0].wt, L[1].wt, L[2].wt, L[0].s, L[1].s, L[2].s,
          L[0].b, L[1].b, L[2].b, L[0].cout, L[1].cout, L[2].cout,
          pts1, 320, coff);
      coff += L[2].cout;
    }
  }

  // ---------------- SA2: FPS 512 -> 128, 3 scales ----------------
  int* c2i = (int*)alloc((size_t)BB * 128 * sizeof(int));
  pn2_fps<<<BB, 256, 0, stream>>>(nx1, 512, 128, c2i);
  float* nx2 = (float*)alloc((size_t)BB * 128 * 3 * sizeof(float));
  pn2_gather_xyz<<<(BB * 128 * 3 + 255) / 256, 256, 0, stream>>>(nx1, c2i, 512,
                                                                 128, nx2);
  h16* pts2 = (h16*)alloc((size_t)BB * 128 * 640 * sizeof(h16));
  {
    const float rad[3] = {0.2f, 0.4f, 0.8f};
    const int   kk[3]  = {32, 64, 128};
    int coff = 0;
    for (int sc = 0; sc < 3; ++sc) {
      int K = kk[sc];
      int* bq = (int*)alloc((size_t)BB * 128 * K * sizeof(int));
      pn2_ball_query<<<(BB * 128) / 8, 256, 0, stream>>>(
          nx1, nx2, 512, 128, K, rad[sc] * rad[sc], bq);
      Lay* L = sa2[sc];
      size_t sh = fusedShmem(K, 352, L[0].cout, L[1].cout, L[2].cout);
      pn2_group_mlp<<<BB * 128, 256, sh, stream>>>(
          nx1, pts1, nx2, bq, 512, 128, K, 320, 352,
          L[0].wt, L[1].wt, L[2].wt, L[0].s, L[1].s, L[2].s,
          L[0].b, L[1].b, L[2].b, L[0].cout, L[1].cout, L[2].cout,
          pts2, 640, coff);
      coff += L[2].cout;
    }
  }

  // ---------------- SA3: group-all MLP 643->256->512->1024 ----------------
  h16* feat3 = (h16*)alloc((size_t)1024 * 672 * sizeof(h16));
  pn2_sa3_feat<<<(1024 * 672 + 255) / 256, 256, 0, stream>>>(nx2, pts2, feat3);
  h16* a1 = (h16*)alloc((size_t)1024 * 256 * sizeof(h16));
  h16* a2 = (h16*)alloc((size_t)1024 * 512 * sizeof(h16));
  h16* a3 = (h16*)alloc((size_t)1024 * 1024 * sizeof(h16));
  {
    int tiles = 64 * 16;
    pn2_gemm16<<<(tiles + 7) / 8, 256, 0, stream>>>(
        feat3, sa3l[0].wt, 672, sa3l[0].s, sa3l[0].b, 1, a1, nullptr, 256,
        64, 16, 1024, 256);
    tiles = 64 * 32;
    pn2_gemm16<<<(tiles + 7) / 8, 256, 0, stream>>>(
        a1, sa3l[1].wt, 256, sa3l[1].s, sa3l[1].b, 1, a2, nullptr, 512,
        64, 32, 1024, 512);
    tiles = 64 * 64;
    pn2_gemm16<<<(tiles + 7) / 8, 256, 0, stream>>>(
        a2, sa3l[2].wt, 512, sa3l[2].s, sa3l[2].b, 1, a3, nullptr, 1024,
        64, 64, 1024, 1024);
  }

  // ---------------- maxpool + FC head ----------------
  h16* fcin = (h16*)alloc((size_t)16 * 1024 * sizeof(h16));
  pn2_sa3_pool<<<(16 * 1024) / 256, 256, 0, stream>>>(a3, fcin);
  h16* f1 = (h16*)alloc((size_t)16 * 512 * sizeof(h16));
  h16* f2 = (h16*)alloc((size_t)16 * 256 * sizeof(h16));
  pn2_gemm16<<<(1 * 32 + 7) / 8, 256, 0, stream>>>(
      fcin, fc[0].wt, 1024, fc[0].s, fc[0].b, 1, f1, nullptr, 512,
      1, 32, 16, 512);
  pn2_gemm16<<<(1 * 16 + 7) / 8, 256, 0, stream>>>(
      f1, fc[1].wt, 512, fc[1].s, fc[1].b, 1, f2, nullptr, 256,
      1, 16, 16, 256);
  // classifier: no scale, no relu, f32 straight to d_out (8 x 40)
  pn2_gemm16<<<1, 256, 0, stream>>>(
      f2, fc[2].wt, 256, nullptr, fc[2].b, 0, nullptr, (float*)d_out, 40,
      1, 3, 8, 40);
}
